// MambaBlock_60327110640123
// MI455X (gfx1250) — matrix operand order repaired and checked by reading, not compiled, not run on hardware
//
#include <hip/hip_runtime.h>
#include <math.h>

// ---------------- types ----------------
typedef _Float16 half_t;
typedef _Float16 v4h  __attribute__((ext_vector_type(4)));
typedef _Float16 v8h  __attribute__((ext_vector_type(8)));
typedef _Float16 v16h __attribute__((ext_vector_type(16)));
typedef float    v8f  __attribute__((ext_vector_type(8)));
// async-to-LDS builtin pointer element types (from hipcc diagnostics):
typedef int v4i_t __attribute__((__vector_size__(16)));
typedef int v2i_t __attribute__((__vector_size__(8)));

#define CAT16(lo, hi) __builtin_shufflevector((lo), (hi), 0,1,2,3,4,5,6,7,8,9,10,11,12,13,14,15)

#define GLOBAL_AS __attribute__((address_space(1)))
#define LDS_AS    __attribute__((address_space(3)))

// CDNA5 async-to-LDS path (ASYNCcnt), with sync fallback if the toolchain
// doesn't declare the builtins.
#if defined(__has_builtin)
#  if __has_builtin(__builtin_amdgcn_global_load_async_to_lds_b128) && \
      __has_builtin(__builtin_amdgcn_global_load_async_to_lds_b64)  && \
      __has_builtin(__builtin_amdgcn_s_wait_asynccnt)
#    define USE_ASYNC 1
#  endif
#endif
#ifndef USE_ASYNC
#  define USE_ASYNC 0
#endif

// ---------------- problem sizes ----------------
#define BATCH   8
#define SEQ     4096
#define DM      768          // d_model == d_inner (EXPAND=1)
#define MTOT    (BATCH*SEQ)  // 32768 rows
#define KDIM    768          // GEMM K for all three GEMMs
#define NXZ     1536
#define DSTATE  16
#define WXCOLS  (DSTATE + DM) // 784
#define NKSTEP  (KDIM / 32)   // 24

// =========================================================================
// prep: a_sum[d] = -sum_n exp(A_log[d,n]);  wB_sum[d] = sum_{n<16} W_x[d,n]
// =========================================================================
__global__ void prep_kernel(const float* __restrict__ A_log,
                            const float* __restrict__ W_x,
                            float* __restrict__ a_sum,
                            float* __restrict__ wB_sum) {
    int d = blockIdx.x * blockDim.x + threadIdx.x;
    if (d >= DM) return;
    float s = 0.f, s2 = 0.f;
    #pragma unroll
    for (int n = 0; n < DSTATE; ++n) {
        s  += __expf(A_log[d * DSTATE + n]);
        s2 += W_x[(size_t)d * WXCOLS + n];
    }
    a_sum[d]  = -s;
    wB_sum[d] = s2;
}

// =========================================================================
// f32 -> f16 weight convert + TRANSPOSE: dst is (Nrows x KDIM) row-major,
// dst[n*KDIM + k] = src[k*ldsrc + coloff + n]
// =========================================================================
__global__ void cvt_t_kernel(const float* __restrict__ src, half_t* __restrict__ dst,
                             int ldsrc, int coloff, int nrows) {
    int i = blockIdx.x * blockDim.x + threadIdx.x;
    if (i < nrows * KDIM) {
        int n = i / KDIM, k = i % KDIM;
        dst[i] = (half_t)src[(size_t)k * ldsrc + coloff + n];
    }
}

// =========================================================================
// LayerNorm: one 256-thread block per row of 768
// =========================================================================
__global__ __launch_bounds__(256)
void ln_kernel(const float* __restrict__ x, const float* __restrict__ gamma,
               const float* __restrict__ beta, half_t* __restrict__ xn) {
    __shared__ float red[256];
    const int m = blockIdx.x, t = threadIdx.x;
    const float* row = x + (size_t)m * DM;
    float v0 = row[t], v1 = row[t + 256], v2 = row[t + 512];
    red[t] = v0 + v1 + v2;
    __syncthreads();
    for (int o = 128; o > 0; o >>= 1) { if (t < o) red[t] += red[t + o]; __syncthreads(); }
    float mu = red[0] * (1.f / DM);
    __syncthreads();
    float d0 = v0 - mu, d1 = v1 - mu, d2 = v2 - mu;
    red[t] = d0 * d0 + d1 * d1 + d2 * d2;
    __syncthreads();
    for (int o = 128; o > 0; o >>= 1) { if (t < o) red[t] += red[t + o]; __syncthreads(); }
    float rstd = rsqrtf(red[0] * (1.f / DM) + 1e-5f);
    half_t* orow = xn + (size_t)m * DM;
    orow[t]       = (half_t)(d0 * rstd * gamma[t]       + beta[t]);
    orow[t + 256] = (half_t)(d1 * rstd * gamma[t + 256] + beta[t + 256]);
    orow[t + 512] = (half_t)(d2 * rstd * gamma[t + 512] + beta[t + 512]);
}

// =========================================================================
// Causal depthwise conv (kernel 4): x1[b,l,d] = sum_j w[d,j]*xz[b,l-3+j,d]
// =========================================================================
__global__ __launch_bounds__(256)
void conv_kernel(const half_t* __restrict__ xzh, const float* __restrict__ convw,
                 half_t* __restrict__ x1h) {
    long idx = (long)blockIdx.x * 256 + threadIdx.x;
    if (idx >= (long)MTOT * DM) return;
    int d = (int)(idx % DM);
    long m = idx / DM;
    int l = (int)(m % SEQ);
    float acc = 0.f;
    #pragma unroll
    for (int j = 0; j < 4; ++j) {
        int lj = l - 3 + j;
        if (lj >= 0) acc += convw[d * 4 + j] * (float)xzh[(size_t)(m - 3 + j) * NXZ + d];
    }
    x1h[(size_t)m * DM + d] = (half_t)acc;
}

// =========================================================================
// Row scalars: hsum[m] = x1.a_sum + (x1.wB_sum)*(x1.1)
// =========================================================================
__global__ __launch_bounds__(256)
void rowscalar_kernel(const half_t* __restrict__ x1h, const float* __restrict__ a_sum,
                      const float* __restrict__ wB_sum, float* __restrict__ hsum) {
    __shared__ float rA[256], rB[256], rS[256];
    const int m = blockIdx.x, t = threadIdx.x;
    float sa = 0.f, sb = 0.f, ss = 0.f;
    for (int c = t; c < DM; c += 256) {
        float v = (float)x1h[(size_t)m * DM + c];
        sa += v * a_sum[c];
        sb += v * wB_sum[c];
        ss += v;
    }
    rA[t] = sa; rB[t] = sb; rS[t] = ss;
    __syncthreads();
    for (int o = 128; o > 0; o >>= 1) {
        if (t < o) { rA[t] += rA[t + o]; rB[t] += rB[t + o]; rS[t] += rS[t + o]; }
        __syncthreads();
    }
    if (t == 0) hsum[m] = rA[0] + rB[0] * rS[0];
}

// =========================================================================
// WMMA GEMM: C(M x N) = A(M x 768) @ Bt^T, A (M x K) f16 row-major,
// Bt is the PRE-TRANSPOSED weight (N x K row-major) so all staging is
// contiguous and async-to-LDS eligible. Block tile 128(M) x 32(N), KT=32,
// double-buffered LDS. 8 waves: 4(M) x 2(N), 2 accumulators each.
// MODE 0: store f16.  MODE 1: fused SSM+silu gate -> yg f16.
// MODE 2: add residual, store f32.
// =========================================================================
template <int MODE>
__global__ __launch_bounds__(256)
void gemm_wmma_kernel(const half_t* __restrict__ A, const half_t* __restrict__ Bt, int N,
                      half_t* __restrict__ outh, float* __restrict__ outf,
                      const float* __restrict__ hsum, const float* __restrict__ Dp,
                      const half_t* __restrict__ x1h, const half_t* __restrict__ xzh,
                      const float* __restrict__ resid) {
    __shared__ half_t lA[2][128][40];   // 32 + 8 pad halfs -> 80B row stride
    __shared__ half_t lB[2][32][40];    // [n][k], weights pre-transposed

    const int t = threadIdx.x;
    const int lane = t & 31, wid = t >> 5;
    const int waveN = wid & 1, waveM = wid >> 1;
    const int mTile = blockIdx.y * 128;
    const int nTile = blockIdx.x * 32;

    v8f acc0 = {}; v8f acc1 = {};

    // staging assignments (per thread, per 32-K stage)
    const int arow = t >> 1;            // 0..127
    const int acol = (t & 1) * 16;      // 0 or 16
    const int brow = t >> 3;            // 0..31  (n within tile)
    const int bc   = (t & 7) * 4;       // 0..28  (k chunk, 4 halfs = 8B)

    const half_t* aSrc = A  + (size_t)(mTile + arow) * KDIM + acol;
    const half_t* bSrc = Bt + (size_t)(nTile + brow) * KDIM + bc;

    auto stage = [&](int kt, int b) {
#if USE_ASYNC
        GLOBAL_AS v4i_t* ga = (GLOBAL_AS v4i_t*)(aSrc + kt);
        LDS_AS char* la = (LDS_AS char*)&lA[b][arow][acol];
        __builtin_amdgcn_global_load_async_to_lds_b128(ga,     (LDS_AS v4i_t*)la,        0, 0);
        __builtin_amdgcn_global_load_async_to_lds_b128(ga + 1, (LDS_AS v4i_t*)(la + 16), 0, 0);
        GLOBAL_AS v2i_t* gb = (GLOBAL_AS v2i_t*)(bSrc + kt);
        __builtin_amdgcn_global_load_async_to_lds_b64(gb, (LDS_AS v2i_t*)&lB[b][brow][bc], 0, 0);
#else
        const v8h* asrc = (const v8h*)(aSrc + kt);
        v8h a0 = asrc[0];
        v8h a1 = asrc[1];
        v4h bv = *(const v4h*)(bSrc + kt);
        *(v8h*)&lA[b][arow][acol]     = a0;
        *(v8h*)&lA[b][arow][acol + 8] = a1;
        *(v4h*)&lB[b][brow][bc]       = bv;
#endif
    };

    // per-lane fragment geometry (ISA 7.12.2 layouts)
    const int rsel = lane & 15;
    const int kb   = (lane >> 4) << 3;  // A: 0 / 8
    const int kb2  = (lane >> 4) << 4;  // unused: 0 / 16

    stage(0, 0);
    int buf = 0;
    for (int ks = 0; ks < NKSTEP; ++ks) {
        if (ks + 1 < NKSTEP) stage((ks + 1) * 32, buf ^ 1);
#if USE_ASYNC
        if (ks + 1 < NKSTEP) __builtin_amdgcn_s_wait_asynccnt(3);  // stage ks resident
        else                 __builtin_amdgcn_s_wait_asynccnt(0);
#endif
        __syncthreads();   // cross-wave visibility of staged tiles

        // B fragment: K = kb..kb+7 and kb+16..kb+23 of column (nTile+waveN*16+rsel)
        v8h b0 = *(const v8h*)&lB[buf][waveN * 16 + rsel][kb];
        v8h b1 = *(const v8h*)&lB[buf][waveN * 16 + rsel][kb + 16];
        v16h bf = CAT16(b0, b1);

        // A fragments: K = kb..kb+7 and kb+16..kb+23 of the lane's row
        v8h a0l = *(const v8h*)&lA[buf][waveM * 32 + rsel][kb];
        v8h a0h = *(const v8h*)&lA[buf][waveM * 32 + rsel][kb + 16];
        v16h af0 = CAT16(a0l, a0h);
        acc0 = __builtin_amdgcn_wmma_f32_16x16x32_f16(false, af0, false, bf,
                                                      (short)0, acc0, false, false);

        v8h a1l = *(const v8h*)&lA[buf][waveM * 32 + 16 + rsel][kb];
        v8h a1h = *(const v8h*)&lA[buf][waveM * 32 + 16 + rsel][kb + 16];
        v16h af1 = CAT16(a1l, a1h);
        acc1 = __builtin_amdgcn_wmma_f32_16x16x32_f16(false, af1, false, bf,
                                                      (short)0, acc1, false, false);

        __syncthreads();   // protect buf^1 before it is restaged next iter
        buf ^= 1;
    }

    // epilogue: C layout — VGPR r, lanes 0-15: M=r, N=lane; lanes 16-31: M=r+8
    const int nG    = nTile + waveN * 16 + rsel;
    const int mhalf = (lane >> 4) * 8;
    #pragma unroll
    for (int s = 0; s < 2; ++s) {
        v8f acc = s ? acc1 : acc0;
        #pragma unroll
        for (int r = 0; r < 8; ++r) {
            const size_t m = (size_t)mTile + waveM * 32 + s * 16 + mhalf + r;
            float v = acc[r];
            if (MODE == 0) {
                outh[m * (size_t)N + nG] = (half_t)v;
            } else if (MODE == 1) {
                float zf  = (float)xzh[m * NXZ + DM + nG];
                float x1v = (float)x1h[m * DM + nG];
                float sil = zf / (1.f + __expf(-zf));
                outh[m * DM + nG] = (half_t)((v * hsum[m] + Dp[nG] * x1v) * sil);
            } else {
                outf[m * DM + nG] = v + resid[m * DM + nG];
            }
        }
    }
}

// =========================================================================
// launcher
// =========================================================================
extern "C" void kernel_launch(void* const* d_in, const int* in_sizes, int n_in,
                              void* d_out, int out_size, void* d_ws, size_t ws_size,
                              hipStream_t stream) {
    const float* x      = (const float*)d_in[0];
    const float* W_in   = (const float*)d_in[1];
    const float* conv_w = (const float*)d_in[2];
    const float* W_x    = (const float*)d_in[3];
    const float* A_log  = (const float*)d_in[4];
    const float* Dp     = (const float*)d_in[5];
    const float* W_out  = (const float*)d_in[6];
    const float* gamma  = (const float*)d_in[7];
    const float* beta   = (const float*)d_in[8];
    float* out = (float*)d_out;

    char* ws = (char*)d_ws;
    size_t off = 0;
    auto alloc = [&](size_t bytes) -> char* {
        off = (off + 255) & ~(size_t)255;
        char* p = ws + off;
        off += bytes;
        return p;
    };

    half_t* xn_h   = (half_t*)alloc((size_t)MTOT * DM * 2);     // reused as yg after GEMM1
    half_t* xz_h   = (half_t*)alloc((size_t)MTOT * NXZ * 2);
    half_t* x1_h   = (half_t*)alloc((size_t)MTOT * DM * 2);
    float*  hsum   = (float*) alloc((size_t)MTOT * 4);
    half_t* Wint   = (half_t*)alloc((size_t)NXZ * KDIM * 2);    // W_in^T  (1536 x 768)
    half_t* Wct    = (half_t*)alloc((size_t)DM * KDIM * 2);     // W_x[:,16:]^T
    half_t* Woutt  = (half_t*)alloc((size_t)DM * KDIM * 2);     // W_out^T
    float*  a_sum  = (float*) alloc(DM * 4);
    float*  wB_sum = (float*) alloc(DM * 4);
    half_t* yg_h   = xn_h;
    (void)in_sizes; (void)n_in; (void)out_size; (void)ws_size; (void)conv_w;

    // 0) tiny prep + transposed weight conversion
    prep_kernel<<<3, 256, 0, stream>>>(A_log, W_x, a_sum, wB_sum);
    cvt_t_kernel<<<(NXZ * KDIM + 255) / 256, 256, 0, stream>>>(W_in,  Wint,  NXZ,    0,      NXZ);
    cvt_t_kernel<<<(DM  * KDIM + 255) / 256, 256, 0, stream>>>(W_x,   Wct,   WXCOLS, DSTATE, DM);
    cvt_t_kernel<<<(DM  * KDIM + 255) / 256, 256, 0, stream>>>(W_out, Woutt, DM,     0,      DM);

    // 1) LayerNorm
    ln_kernel<<<MTOT, 256, 0, stream>>>(x, gamma, beta, xn_h);

    // 2) GEMM1: xz = xn @ W_in   (M x 1536)
    gemm_wmma_kernel<0><<<dim3(NXZ / 32, MTOT / 128), 256, 0, stream>>>(
        xn_h, Wint, NXZ, xz_h, nullptr, nullptr, nullptr, nullptr, nullptr, nullptr);

    // 3) causal depthwise conv -> x1
    conv_kernel<<<(int)(((long)MTOT * DM + 255) / 256), 256, 0, stream>>>(xz_h, conv_w, x1_h);

    // 4) per-row scalars hsum
    rowscalar_kernel<<<MTOT, 256, 0, stream>>>(x1_h, a_sum, wB_sum, hsum);

    // 5) GEMM2: Cm = x1 @ Wc, fused (Cm*hsum + Dp*x1)*silu(z) -> yg
    gemm_wmma_kernel<1><<<dim3(DM / 32, MTOT / 128), 256, 0, stream>>>(
        x1_h, Wct, DM, yg_h, nullptr, hsum, Dp, x1_h, xz_h, nullptr);

    // 6) GEMM3: out = yg @ W_out + residual
    gemm_wmma_kernel<2><<<dim3(DM / 32, MTOT / 128), 256, 0, stream>>>(
        yg_h, Woutt, DM, nullptr, out, nullptr, nullptr, nullptr, nullptr, x);
}
